// GCN_19791209300130
// MI455X (gfx1250) — matrix operand order repaired and checked by reading, not compiled, not run on hardware
//
#include <hip/hip_runtime.h>
#include <math.h>

#define FEAT   1024
#define NNODE  4096
#define NEDGE  4096
#define NNZT   65536
#define HIDD   512
#define CATF   2048   // FEAT + 2*HIDD

typedef __attribute__((ext_vector_type(16))) __bf16 v16bf;
typedef __attribute__((ext_vector_type(8)))  float  v8f;

// ---------------------------------------------------------------- helpers
__device__ __forceinline__ unsigned short f2bf(float f) {
  unsigned int u = __float_as_uint(f);
  u += 0x7FFFu + ((u >> 16) & 1u);   // round-to-nearest-even
  return (unsigned short)(u >> 16);
}

__device__ __forceinline__ void atomicMaxF(float* addr, float val) {
  if (val >= 0.f) atomicMax((int*)addr, __float_as_int(val));
  else            atomicMin((unsigned int*)addr, __float_as_uint(val));
}

// ---------------------------------------------------------------- WMMA GEMM
// C[M,N] = act( sum_k A[m*sa_m + k*sa_k] * B[n*sb_n + k] + bias[n] )
// REQUIRES: M%128==0, N%64==0, K%32==0 (true at every call site).
// Block: 256 threads = 8 waves; block tile 128x64; wave tile 32x32 (4 WMMA accs).
// LDS double-buffered (ping-pong) -> one barrier per K-step.
__global__ __launch_bounds__(256)
void k_gemm_bf16(const float* __restrict__ A, long long sa_m, long long sa_k,
                 const float* __restrict__ B, long long sb_n,
                 float* __restrict__ C, const float* __restrict__ bias,
                 int M, int N, int K, int act)
{
  __shared__ unsigned short sA[2 * 128 * 32];   // [buf][m][k] bf16, 2x8 KB
  __shared__ unsigned short sB[2 * 64 * 32];    // [buf][n][k] bf16, 2x4 KB

  const int tid  = threadIdx.x;
  const int lane = tid & 31;
  const int wv   = tid >> 5;                // 0..7
  const int wr   = wv & 3;                  // wave row (M dir)
  const int wc   = wv >> 2;                 // wave col (N dir)
  const int m0   = blockIdx.y * 128;
  const int n0   = blockIdx.x * 64;

  // cooperative-load mapping: 8 quads of 4 k-values x 32 rows
  const int q4 = (tid & 7) * 4;             // k offset within tile (0,4,..,28)
  const int r0 = tid >> 3;                  // row 0..31

  v8f acc[2][2] = {};

  const int hsel  = lane >> 4;              // half-wave select
  const int l16   = lane & 15;
  const int abase = hsel * 8;               // A frag K base (ISA 16-bit A layout)
  const int bbase = hsel * 8;               // B frag K base

  const bool a_rowmajor = (sa_k == 1);

  for (int k0 = 0; k0 < K; k0 += 32) {
    unsigned short* tA = sA + ((k0 >> 5) & 1) * (128 * 32);
    unsigned short* tB = sB + ((k0 >> 5) & 1) * (64 * 32);

    if (a_rowmajor) {
      #pragma unroll
      for (int i = 0; i < 4; ++i) {         // A tile 128x32 (4 rows/thread)
        int r = r0 + 32 * i;
        const float4 v = *(const float4*)&A[(long long)(m0 + r) * sa_m + k0 + q4];
        ushort4 h;
        h.x = f2bf(v.x); h.y = f2bf(v.y); h.z = f2bf(v.z); h.w = f2bf(v.w);
        *(ushort4*)&tA[r * 32 + q4] = h;
      }
      if (k0 + 32 < K)
        __builtin_prefetch(&A[(long long)(m0 + r0) * sa_m + k0 + 32 + q4], 0, 1);
    } else {                                // strided-A (transposed view)
      #pragma unroll
      for (int i = 0; i < 4; ++i) {
        int r = r0 + 32 * i;
        const float* ap = &A[(long long)(m0 + r) * sa_m + (long long)(k0 + q4) * sa_k];
        ushort4 h;
        h.x = f2bf(ap[0]);
        h.y = f2bf(ap[sa_k]);
        h.z = f2bf(ap[2 * sa_k]);
        h.w = f2bf(ap[3 * sa_k]);
        *(ushort4*)&tA[r * 32 + q4] = h;
      }
    }
    #pragma unroll
    for (int i = 0; i < 2; ++i) {           // B tile 64x32 (2 rows/thread)
      int r = r0 + 32 * i;
      const float4 v = *(const float4*)&B[(long long)(n0 + r) * sb_n + k0 + q4];
      ushort4 h;
      h.x = f2bf(v.x); h.y = f2bf(v.y); h.z = f2bf(v.z); h.w = f2bf(v.w);
      *(ushort4*)&tB[r * 32 + q4] = h;
    }
    __syncthreads();

    union Frag { v16bf v; uint4 q[2]; };
    Frag fb[2];
    #pragma unroll
    for (int ni = 0; ni < 2; ++ni) {
      int nn = (wc * 32 + 16 * ni + l16) * 32;
      fb[ni].q[0] = *(const uint4*)&tB[nn + bbase];
      fb[ni].q[1] = *(const uint4*)&tB[nn + 16 + bbase];
    }
    #pragma unroll
    for (int mi = 0; mi < 2; ++mi) {
      Frag fa;
      int mm = (wr * 32 + 16 * mi + l16) * 32;
      fa.q[0] = *(const uint4*)&tA[mm + abase];       // K = abase..abase+7
      fa.q[1] = *(const uint4*)&tA[mm + 16 + abase];  // K = 16+abase..
      #pragma unroll
      for (int ni = 0; ni < 2; ++ni) {
        acc[mi][ni] = __builtin_amdgcn_wmma_f32_16x16x32_bf16(
            false, fa.v, false, fb[ni].v, (short)0, acc[mi][ni], false, false);
      }
    }
  }

  // epilogue: C/D layout — VGPR r holds row r (lanes 0-15) / row 8+r (lanes 16-31)
  #pragma unroll
  for (int mi = 0; mi < 2; ++mi) {
    #pragma unroll
    for (int ni = 0; ni < 2; ++ni) {
      int rbase = m0 + wr * 32 + 16 * mi + 8 * hsel;
      int col   = n0 + wc * 32 + 16 * ni + l16;
      float bv  = (bias != nullptr) ? bias[col] : 0.f;
      #pragma unroll
      for (int r = 0; r < 8; ++r) {
        float v = acc[mi][ni][r] + bv;
        if (act == 1)      v = v > 0.f  ? v : 0.f;        // relu
        else if (act == 2) v = v >= 0.f ? v : 0.01f * v;  // leaky 0.01
        C[(long long)(rbase + r) * N + col] = v;
      }
    }
  }
}

// ---------------------------------------------------------------- small kernels
__global__ void k_fill(float* p, float v, int n) {
  int i = blockIdx.x * blockDim.x + threadIdx.x;
  if (i < n) p[i] = v;
}

__global__ void k_inv(float* p, int n) {
  int i = blockIdx.x * blockDim.x + threadIdx.x;
  if (i < n) p[i] = (p[i] > 0.f) ? 1.f / p[i] : 0.f;
}

__global__ void k_deg(const int* __restrict__ row, const int* __restrict__ col,
                      float* D, float* Bc, int nnz) {
  int i = blockIdx.x * blockDim.x + threadIdx.x;
  if (i < nnz) { atomicAdd(&D[row[i]], 1.f); atomicAdd(&Bc[col[i]], 1.f); }
}

// out[r] = dot(X[r,0:F], v)  — one wave per row
__global__ __launch_bounds__(256)
void k_rowdot(const float* __restrict__ X, const float* __restrict__ v,
              float* __restrict__ out, int rows, int F) {
  int w = (blockIdx.x * blockDim.x + threadIdx.x) >> 5;
  int lane = threadIdx.x & 31;
  if (w >= rows) return;
  const float* xr = X + (long long)w * F;
  float s = 0.f;
  for (int k = lane; k < F; k += 32) s += xr[k] * v[k];
  for (int off = 16; off; off >>= 1) s += __shfl_down(s, off, 32);
  if (lane == 0) out[w] = s;
}

__global__ void k_inc_logit(const int* __restrict__ row, const int* __restrict__ col,
                            const float* __restrict__ xn, const float* __restrict__ en,
                            float* __restrict__ a, int nnz) {
  int i = blockIdx.x * blockDim.x + threadIdx.x;
  if (i < nnz) {
    float v = xn[row[i]] + en[col[i]];
    a[i] = v >= 0.f ? v : 0.2f * v;           // leaky 0.2
  }
}

__global__ void k_segmax(const int* __restrict__ col, const float* __restrict__ a,
                         float* amax, int nnz) {
  int i = blockIdx.x * blockDim.x + threadIdx.x;
  if (i < nnz) atomicMaxF(&amax[col[i]], a[i]);
}

__global__ void k_expsum(const int* __restrict__ col, float* __restrict__ a,
                         const float* __restrict__ amax, float* denom, int nnz) {
  int i = blockIdx.x * blockDim.x + threadIdx.x;
  if (i < nnz) {
    float e = __expf(a[i] - amax[col[i]]);
    a[i] = e;
    atomicAdd(&denom[col[i]], e);
  }
}

__global__ void k_alpha(const int* __restrict__ col, float* __restrict__ a,
                        const float* __restrict__ denom, int nnz) {
  int i = blockIdx.x * blockDim.x + threadIdx.x;
  if (i < nnz) a[i] = a[i] / (denom[col[i]] + 1e-16f);
}

// dst[didx[k], :] += alpha[k]*wdeg[widx[k]] * src[sidx[k], :]   — one wave per incidence
__global__ __launch_bounds__(256)
void k_scatter(const int* __restrict__ sidx, const int* __restrict__ didx,
               const int* __restrict__ widx, const float* __restrict__ alpha,
               const float* __restrict__ wdeg, const float* __restrict__ src,
               float* __restrict__ dst, int nnz, int F) {
  int w = (blockIdx.x * blockDim.x + threadIdx.x) >> 5;
  int lane = threadIdx.x & 31;
  if (w >= nnz) return;
  float c = alpha[w] * wdeg[widx[w]];
  const float* s = src + (long long)sidx[w] * F;
  float* d = dst + (long long)didx[w] * F;
  for (int f = lane; f < F; f += 32) atomicAdd(&d[f], c * s[f]);
}

// column stats of (h + hb): one thread per column, rows strided (coalesced)
__global__ void k_colstats(const float* __restrict__ h, const float* __restrict__ hb,
                           const float* __restrict__ ms, float* mean, float* rstd,
                           int rows, int F) {
  int j = blockIdx.x * blockDim.x + threadIdx.x;
  if (j >= F) return;
  float bj = hb[j], s1 = 0.f, s2 = 0.f;
  for (int i = 0; i < rows; ++i) {
    float v = h[(long long)i * F + j] + bj;
    s1 += v; s2 += v * v;
  }
  float mu = s1 / rows, m = ms[j];
  float var = s2 / rows - 2.f * m * mu * mu + m * m * mu * mu;
  mean[j] = mu;
  rstd[j] = rsqrtf(var + 1e-5f);
}

__global__ void k_gnapply(float* __restrict__ h, const float* __restrict__ hb,
                          const float* __restrict__ gw, const float* __restrict__ gb,
                          const float* __restrict__ ms, const float* __restrict__ mean,
                          const float* __restrict__ rstd, long long n, int F) {
  long long i = (long long)blockIdx.x * blockDim.x + threadIdx.x;
  if (i >= n) return;
  int j = (int)(i % F);
  float v = h[i] + hb[j];
  float y = gw[j] * (v - mean[j] * ms[j]) * rstd[j] + gb[j];
  h[i] = y >= 0.f ? y : 0.01f * y;            // leaky 0.01
}

__global__ void k_cat(const float* __restrict__ x, const float* __restrict__ o1,
                      const float* __restrict__ o2, float* __restrict__ cat, long long n) {
  long long i = (long long)blockIdx.x * blockDim.x + threadIdx.x;
  if (i >= n) return;
  int r = (int)(i / CATF), c = (int)(i % CATF);
  float v;
  if      (c < FEAT)        v = x [(long long)r * FEAT + c];
  else if (c < FEAT + HIDD) v = o1[(long long)r * HIDD + (c - FEAT)];
  else                      v = o2[(long long)r * HIDD + (c - FEAT - HIDD)];
  cat[i] = v;
}

// s[r] = sigmoid(dot(T[r,:], w) + b)
__global__ __launch_bounds__(256)
void k_srow(const float* __restrict__ T, const float* __restrict__ w,
            const float* __restrict__ b, float* __restrict__ s, int rows, int K) {
  int r = (blockIdx.x * blockDim.x + threadIdx.x) >> 5;
  int lane = threadIdx.x & 31;
  if (r >= rows) return;
  const float* tr = T + (long long)r * K;
  float acc = 0.f;
  for (int k = lane; k < K; k += 32) acc += tr[k] * w[k];
  for (int off = 16; off; off >>= 1) acc += __shfl_down(acc, off, 32);
  if (lane == 0) s[r] = 1.f / (1.f + __expf(-(acc + b[0])));
}

__global__ void k_sfinal(float* s, int n) {   // launch single block of 1024
  __shared__ float red[1024];
  float loc = 0.f;
  for (int i = threadIdx.x; i < n; i += blockDim.x) loc += s[i];
  red[threadIdx.x] = loc;
  __syncthreads();
  for (int off = blockDim.x >> 1; off; off >>= 1) {
    if ((int)threadIdx.x < off) red[threadIdx.x] += red[threadIdx.x + off];
    __syncthreads();
  }
  float mu = red[0] / n;
  for (int i = threadIdx.x; i < n; i += blockDim.x) s[i] -= mu;
}

// logits[r, 0:4] = sum_k cat[r,k]*s[k]*W[o,k] + b[o]   — one wave per row
__global__ __launch_bounds__(256)
void k_logits(const float* __restrict__ cat, const float* __restrict__ s,
              const float* __restrict__ W, const float* __restrict__ b,
              float* __restrict__ out, int rows, int K) {
  int r = (blockIdx.x * blockDim.x + threadIdx.x) >> 5;
  int lane = threadIdx.x & 31;
  if (r >= rows) return;
  const float* cr = cat + (long long)r * K;
  float a0 = 0.f, a1 = 0.f, a2 = 0.f, a3 = 0.f;
  for (int k = lane; k < K; k += 32) {
    float v = cr[k] * s[k];
    a0 += v * W[0 * K + k]; a1 += v * W[1 * K + k];
    a2 += v * W[2 * K + k]; a3 += v * W[3 * K + k];
  }
  for (int off = 16; off; off >>= 1) {
    a0 += __shfl_down(a0, off, 32); a1 += __shfl_down(a1, off, 32);
    a2 += __shfl_down(a2, off, 32); a3 += __shfl_down(a3, off, 32);
  }
  if (lane == 0) {
    out[(long long)r * 4 + 0] = a0 + b[0];
    out[(long long)r * 4 + 1] = a1 + b[1];
    out[(long long)r * 4 + 2] = a2 + b[2];
    out[(long long)r * 4 + 3] = a3 + b[3];
  }
}

// ---------------------------------------------------------------- host side
static void launch_gemm(const float* A, long long sam, long long sak,
                        const float* B, long long sbn,
                        float* C, const float* bias, int M, int N, int K, int act,
                        hipStream_t s) {
  dim3 g((N + 63) / 64, (M + 127) / 128);
  k_gemm_bf16<<<g, 256, 0, s>>>(A, sam, sak, B, sbn, C, bias, M, N, K, act);
}

static void run_hg_layer(const float* xin, const float* eattr,
                         const float* W, const float* att,
                         float* XW, float* EW, float* MB, float* Hout,
                         float* xn, float* en, float* amax, float* den, float* a,
                         const float* Dinv, const float* Binv,
                         const int* erow, const int* ecol, hipStream_t st) {
  launch_gemm(xin,  FEAT, 1, W, FEAT, XW, nullptr, NNODE, FEAT, FEAT, 0, st);
  launch_gemm(eattr, FEAT, 1, W, FEAT, EW, nullptr, NEDGE, FEAT, FEAT, 0, st);
  k_rowdot<<<(NNODE * 32 + 255) / 256, 256, 0, st>>>(XW, att,        xn, NNODE, FEAT);
  k_rowdot<<<(NEDGE * 32 + 255) / 256, 256, 0, st>>>(EW, att + FEAT, en, NEDGE, FEAT);
  k_inc_logit<<<NNZT / 256, 256, 0, st>>>(erow, ecol, xn, en, a, NNZT);
  k_fill<<<(NEDGE + 255) / 256, 256, 0, st>>>(amax, -INFINITY, NEDGE);
  k_segmax<<<NNZT / 256, 256, 0, st>>>(ecol, a, amax, NNZT);
  hipMemsetAsync(den, 0, NEDGE * sizeof(float), st);
  k_expsum<<<NNZT / 256, 256, 0, st>>>(ecol, a, amax, den, NNZT);
  k_alpha<<<NNZT / 256, 256, 0, st>>>(ecol, a, den, NNZT);
  hipMemsetAsync(MB, 0, (size_t)NEDGE * FEAT * sizeof(float), st);
  k_scatter<<<NNZT * 32 / 256, 256, 0, st>>>(erow, ecol, ecol, a, Binv, XW, MB, NNZT, FEAT);
  hipMemsetAsync(Hout, 0, (size_t)NNODE * FEAT * sizeof(float), st);
  k_scatter<<<NNZT * 32 / 256, 256, 0, st>>>(ecol, erow, erow, a, Dinv, MB, Hout, NNZT, FEAT);
}

extern "C" void kernel_launch(void* const* d_in, const int* in_sizes, int n_in,
                              void* d_out, int out_size, void* d_ws, size_t ws_size,
                              hipStream_t stream) {
  (void)in_sizes; (void)n_in; (void)out_size; (void)ws_size;
  const float* x       = (const float*)d_in[0];
  const int*   eidx    = (const int*)d_in[1];
  const int*   erow    = eidx;          // edge_index[0] : node ids
  const int*   ecol    = eidx + NNZT;   // edge_index[1] : hyperedge ids
  const float* eattr   = (const float*)d_in[2];
  const float* hg1_W   = (const float*)d_in[3];
  const float* hg1_att = (const float*)d_in[4];
  const float* hg1_b   = (const float*)d_in[5];
  const float* hg2_W   = (const float*)d_in[6];
  const float* hg2_att = (const float*)d_in[7];
  const float* hg2_b   = (const float*)d_in[8];
  const float* gn1_w   = (const float*)d_in[9];
  const float* gn1_b   = (const float*)d_in[10];
  const float* gn1_ms  = (const float*)d_in[11];
  const float* gn2_w   = (const float*)d_in[12];
  const float* gn2_b   = (const float*)d_in[13];
  const float* gn2_ms  = (const float*)d_in[14];
  const float* fc1_W   = (const float*)d_in[15];
  const float* fc1_b   = (const float*)d_in[16];
  const float* fc2_W   = (const float*)d_in[17];
  const float* fc2_b   = (const float*)d_in[18];
  const float* att1_W  = (const float*)d_in[19];
  const float* att1_b  = (const float*)d_in[20];
  const float* att2_W  = (const float*)d_in[21];
  const float* att2_b  = (const float*)d_in[22];
  const float* cls_W   = (const float*)d_in[23];
  const float* cls_b   = (const float*)d_in[24];
  float* out = (float*)d_out;

  // ---- workspace layout (floats); CAT aliases XW+EW, T aliases MB+H1
  float* ws = (float*)d_ws;
  size_t o = 0;
  float* XW = ws + o; o += (size_t)NNODE * FEAT;   // 4 M
  float* EW = ws + o; o += (size_t)NEDGE * FEAT;   // 4 M   (also H2)
  float* MB = ws + o; o += (size_t)NEDGE * FEAT;   // 4 M
  float* H1 = ws + o; o += (size_t)NNODE * FEAT;   // 4 M
  float* O1 = ws + o; o += (size_t)NNODE * HIDD;   // 2 M
  float* O2 = ws + o; o += (size_t)NNODE * HIDD;   // 2 M
  float* A_   = ws + o; o += NNZT;
  float* XN   = ws + o; o += NNODE;
  float* EN   = ws + o; o += NEDGE;
  float* AMAX = ws + o; o += NEDGE;
  float* DEN  = ws + o; o += NEDGE;
  float* DINV = ws + o; o += NNODE;
  float* BINV = ws + o; o += NEDGE;
  float* MEAN = ws + o; o += FEAT;
  float* RSTD = ws + o; o += FEAT;
  float* SB   = ws + o; o += CATF;
  float* CAT = XW;   // 8 M contiguous (XW+EW), used after layer 2 done
  float* T   = MB;   // 8 M contiguous (MB+H1), used after CAT built

  // ---- degree normalizations (shared by both layers)
  hipMemsetAsync(DINV, 0, NNODE * sizeof(float), stream);
  hipMemsetAsync(BINV, 0, NEDGE * sizeof(float), stream);
  k_deg<<<NNZT / 256, 256, 0, stream>>>(erow, ecol, DINV, BINV, NNZT);
  k_inv<<<(NNODE + 255) / 256, 256, 0, stream>>>(DINV, NNODE);
  k_inv<<<(NEDGE + 255) / 256, 256, 0, stream>>>(BINV, NEDGE);

  const long long nF = (long long)NNODE * FEAT;

  // ---- layer 1
  run_hg_layer(x, eattr, hg1_W, hg1_att, XW, EW, MB, H1,
               XN, EN, AMAX, DEN, A_, DINV, BINV, erow, ecol, stream);
  k_colstats<<<FEAT / 256, 256, 0, stream>>>(H1, hg1_b, gn1_ms, MEAN, RSTD, NNODE, FEAT);
  k_gnapply<<<(int)((nF + 255) / 256), 256, 0, stream>>>(H1, hg1_b, gn1_w, gn1_b,
                                                         gn1_ms, MEAN, RSTD, nF, FEAT);
  launch_gemm(H1, FEAT, 1, fc1_W, FEAT, O1, fc1_b, NNODE, HIDD, FEAT, 2, stream);

  // ---- layer 2 (input = H1, output lands in EW)
  run_hg_layer(H1, eattr, hg2_W, hg2_att, XW, EW, MB, EW,
               XN, EN, AMAX, DEN, A_, DINV, BINV, erow, ecol, stream);
  k_colstats<<<FEAT / 256, 256, 0, stream>>>(EW, hg2_b, gn2_ms, MEAN, RSTD, NNODE, FEAT);
  k_gnapply<<<(int)((nF + 255) / 256), 256, 0, stream>>>(EW, hg2_b, gn2_w, gn2_b,
                                                         gn2_ms, MEAN, RSTD, nF, FEAT);
  launch_gemm(EW, FEAT, 1, fc2_W, FEAT, O2, fc2_b, NNODE, HIDD, FEAT, 2, stream);

  // ---- concat [x, O1, O2] -> CAT[4096, 2048]  (overwrites XW+EW region)
  const long long nC = (long long)NNODE * CATF;
  k_cat<<<(int)((nC + 255) / 256), 256, 0, stream>>>(x, O1, O2, CAT, nC);

  // ---- T[2048,4096] = relu(CATᵀ @ att1_Wᵀ + att1_b)   (A = CAT transposed view)
  launch_gemm(CAT, 1, CATF, att1_W, NNODE, T, att1_b, CATF, NNODE, NNODE, 1, stream);

  // ---- s = sigmoid(T @ att2_Wᵀ + att2_b); s -= mean(s)
  k_srow<<<(CATF * 32 + 255) / 256, 256, 0, stream>>>(T, att2_W, att2_b, SB, CATF, NNODE);
  k_sfinal<<<1, 1024, 0, stream>>>(SB, CATF);

  // ---- logits = (CAT * s) @ cls_Wᵀ + cls_b
  k_logits<<<(NNODE * 32 + 255) / 256, 256, 0, stream>>>(CAT, SB, cls_W, cls_b, out,
                                                         NNODE, CATF);
}
